// ScaledDotProductAttention_10892037063261
// MI455X (gfx1250) — matrix operand order repaired and checked by reading, not compiled, not run on hardware
//
#include <hip/hip_runtime.h>
#include <stdint.h>

typedef __attribute__((ext_vector_type(16))) __bf16 v16bf;
typedef __attribute__((ext_vector_type(8)))  float v8f;

// Problem constants (B=2, H=16, S=2048, DK=DV=64)
constexpr int SEQ  = 2048;
constexpr int DKC  = 64;
constexpr int DVC  = 64;
constexpr int BH   = 32;            // B*H
constexpr int NKT  = SEQ / 16;      // 128 key tiles
constexpr float SCALE = 0.125f;     // 1/sqrt(64)

__global__ __launch_bounds__(256) void sdpa_wmma_kernel(
    const float* __restrict__ Q, const float* __restrict__ K,
    const float* __restrict__ V, const uint8_t* __restrict__ mask,
    float* __restrict__ ctx_out, float* __restrict__ attn_out)
{
    // per-wave bf16 staging tile for P (16 rows x 32 keys)
    __shared__ __bf16 pstage[8][16 * 32];

    const int tid   = threadIdx.x;
    const int lane  = tid & 31;
    const int wslot = tid >> 5;
    const int wave  = blockIdx.x * 8 + wslot;   // 0 .. 4095
    const int bh    = wave >> 7;                // wave / 128
    const int qt    = wave & 127;               // query tile within (b,h)

    const float*   Qb = Q + ((size_t)bh * SEQ + qt * 16) * DKC;
    const float*   Kb = K + (size_t)bh * SEQ * DKC;
    const float*   Vb = V + (size_t)bh * SEQ * DVC;
    const uint8_t* Mb = mask + ((size_t)bh * SEQ + qt * 16) * SEQ;
    float*         Ab = attn_out + ((size_t)bh * SEQ + qt * 16) * SEQ;
    float*         Cb = ctx_out  + ((size_t)bh * SEQ + qt * 16) * DVC;

    const int row = lane & 15;   // A-frag / C-frag indexing
    const int hi  = lane >> 4;
    const int col = lane & 15;   // B-frag / C-frag column (N)
    const int kb  = hi * 16;     // B-frag K base for this half

    // ---------------- Q A-fragments (bf16), kept resident ----------------
    v16bf a0, a1;
    {
        const float* qrow = Qb + row * DKC;
        #pragma unroll
        for (int v = 0; v < 8; ++v) {
            const int k = 16 * hi + 2 * v;
            a0[2 * v]     = (__bf16)qrow[k];
            a0[2 * v + 1] = (__bf16)qrow[k + 1];
            a1[2 * v]     = (__bf16)qrow[32 + k];
            a1[2 * v + 1] = (__bf16)qrow[32 + k + 1];
        }
    }

    // ---------------- Pass 1: online softmax statistics ----------------
    float m[8], l[8];
    #pragma unroll
    for (int v = 0; v < 8; ++v) { m[v] = -INFINITY; l[v] = 0.0f; }

    for (int j = 0; j < NKT; ++j) {
        const float* krow = Kb + (size_t)(j * 16 + col) * DKC;
        v16bf b0, b1;
        #pragma unroll
        for (int v = 0; v < 8; ++v) {
            b0[2 * v]     = (__bf16)krow[kb + 2 * v];
            b0[2 * v + 1] = (__bf16)krow[kb + 2 * v + 1];
            b1[2 * v]     = (__bf16)krow[32 + kb + 2 * v];
            b1[2 * v + 1] = (__bf16)krow[32 + kb + 2 * v + 1];
        }
        v8f c = {};
        c = __builtin_amdgcn_wmma_f32_16x16x32_bf16(false, a0, false, b0,
                                                    (short)0, c, false, false);
        c = __builtin_amdgcn_wmma_f32_16x16x32_bf16(false, a1, false, b1,
                                                    (short)0, c, false, false);
        #pragma unroll
        for (int v = 0; v < 8; ++v) {
            const int r = v + 8 * hi;
            const float s = Mb[(size_t)r * SEQ + j * 16 + col] ? -1e9f
                                                               : c[v] * SCALE;
            const float mn = fmaxf(m[v], s);
            l[v] = l[v] * __expf(m[v] - mn) + __expf(s - mn);
            m[v] = mn;
        }
    }

    // merge (m,l) across the 16 lanes that share each row (xor<16 stays in half)
    #pragma unroll
    for (int off = 1; off < 16; off <<= 1) {
        #pragma unroll
        for (int v = 0; v < 8; ++v) {
            const float m2 = __shfl_xor(m[v], off, 32);
            const float l2 = __shfl_xor(l[v], off, 32);
            const float mn = fmaxf(m[v], m2);
            l[v] = l[v] * __expf(m[v] - mn) + l2 * __expf(m2 - mn);
            m[v] = mn;
        }
    }
    float rl[8];
    #pragma unroll
    for (int v = 0; v < 8; ++v) rl[v] = __frcp_rn(l[v]);

    // ---------------- Pass 2: emit attn + accumulate context ----------------
    v8f acc0 = {}, acc1 = {}, acc2 = {}, acc3 = {};
    __bf16* ps = pstage[wslot];

    for (int t = 0; t < NKT / 2; ++t) {       // 32 keys per chunk
        #pragma unroll
        for (int jj = 0; jj < 2; ++jj) {
            const int j = 2 * t + jj;
            const float* krow = Kb + (size_t)(j * 16 + col) * DKC;
            v16bf b0, b1;
            #pragma unroll
            for (int v = 0; v < 8; ++v) {
                b0[2 * v]     = (__bf16)krow[kb + 2 * v];
                b0[2 * v + 1] = (__bf16)krow[kb + 2 * v + 1];
                b1[2 * v]     = (__bf16)krow[32 + kb + 2 * v];
                b1[2 * v + 1] = (__bf16)krow[32 + kb + 2 * v + 1];
            }
            v8f c = {};
            c = __builtin_amdgcn_wmma_f32_16x16x32_bf16(false, a0, false, b0,
                                                        (short)0, c, false, false);
            c = __builtin_amdgcn_wmma_f32_16x16x32_bf16(false, a1, false, b1,
                                                        (short)0, c, false, false);
            #pragma unroll
            for (int v = 0; v < 8; ++v) {
                const int r = v + 8 * hi;
                const float s = Mb[(size_t)r * SEQ + j * 16 + col] ? -1e9f
                                                                   : c[v] * SCALE;
                const float p = __expf(s - m[v]) * rl[v];
                Ab[(size_t)r * SEQ + j * 16 + col] = p;          // normalized attn
                ps[r * 32 + jj * 16 + col] = (__bf16)p;          // stage for GEMM2
            }
        }

        // P A-fragment (16 rows x 32 keys) from LDS (C-layout -> A-layout)
        v16bf pa;
        #pragma unroll
        for (int v = 0; v < 8; ++v) {
            const int k = 16 * hi + 2 * v;
            pa[2 * v]     = ps[row * 32 + k];
            pa[2 * v + 1] = ps[row * 32 + k + 1];
        }

        // V B-fragments: keys t*32 + [0..31], 4 N-tiles of 16 (DV = 64)
        const float* vbase = Vb + (size_t)(t * 32 + kb) * DVC;
        #pragma unroll
        for (int d = 0; d < 4; ++d) {
            v16bf vb;
            #pragma unroll
            for (int v = 0; v < 8; ++v) {
                vb[2 * v]     = (__bf16)vbase[(size_t)(2 * v) * DVC + d * 16 + col];
                vb[2 * v + 1] = (__bf16)vbase[(size_t)(2 * v + 1) * DVC + d * 16 + col];
            }
            v8f* ap = (d == 0) ? &acc0 : (d == 1) ? &acc1 : (d == 2) ? &acc2 : &acc3;
            *ap = __builtin_amdgcn_wmma_f32_16x16x32_bf16(false, pa, false, vb,
                                                          (short)0, *ap, false, false);
        }
    }

    // ---------------- write context ----------------
    #pragma unroll
    for (int v = 0; v < 8; ++v) {
        const int r = v + 8 * hi;
        Cb[(size_t)r * DVC + 0 * 16 + col]  = acc0[v];
        Cb[(size_t)r * DVC + 1 * 16 + col]  = acc1[v];
        Cb[(size_t)r * DVC + 2 * 16 + col]  = acc2[v];
        Cb[(size_t)r * DVC + 3 * 16 + col]  = acc3[v];
    }
}

extern "C" void kernel_launch(void* const* d_in, const int* in_sizes, int n_in,
                              void* d_out, int out_size, void* d_ws, size_t ws_size,
                              hipStream_t stream) {
    (void)in_sizes; (void)n_in; (void)out_size; (void)d_ws; (void)ws_size;
    const float*   Q  = (const float*)d_in[0];
    const float*   K  = (const float*)d_in[1];
    const float*   V  = (const float*)d_in[2];
    const uint8_t* Mk = (const uint8_t*)d_in[3];   // numpy bool = 1 byte

    float* out  = (float*)d_out;
    float* ctx  = out;                                   // [B,H,S,DV] first
    float* attn = out + (size_t)BH * SEQ * DVC;          // then [B,H,S,S]

    // 4096 query tiles, 8 waves (256 threads) per block -> 512 blocks
    sdpa_wmma_kernel<<<512, 256, 0, stream>>>(Q, K, V, Mk, ctx, attn);
}
